// SipMaskInference_85212151153057
// MI455X (gfx1250) — compile-verified, matrix operand order checked by reading, not run on hardware
//
#include <hip/hip_runtime.h>
#include <stdint.h>

typedef unsigned int u32;
typedef unsigned long long u64;

#define NLVL 3
#define NIMG 8
#define NCLS 80
#define COFD 128
#define BINS 4096
#define CAP  4096
#define TOPK 1000
#define KTOT 3000
#define PADK 3072
#define SORTN 4096
#define POST 100
#define IMGW 1216.0f
#define IMGH 800.0f
#define CLSOFF 1217.0f
#define PRETH 0.05f
#define NMSTH 0.6f

// workspace offsets (bytes)
#define WS_HIST 0
#define WS_META (24 * BINS * 4)
#define WS_CBUF (WS_META + 512)
#define WS_REC  (WS_CBUF + 24 * CAP * 8)
#define WS_SREC (WS_REC  + NIMG * PADK * 32)
#define WS_NMSB (WS_SREC + NIMG * PADK * 32)
#define WS_IDX  (WS_NMSB + NIMG * PADK * 16)

// output offsets (floats): boxes, scores, cls, cofs, kv
#define O_BOX 0
#define O_SCR (NIMG * POST * 4)
#define O_CLS (O_SCR + NIMG * POST)
#define O_COF (O_CLS + NIMG * POST)
#define O_KV  (O_COF + NIMG * POST * COFD)
#define O_TOT (O_KV  + NIMG * POST)

typedef __attribute__((ext_vector_type(16))) _Float16 v16h;
typedef __attribute__((ext_vector_type(8)))  float    v8f;
typedef __attribute__((ext_vector_type(4)))  u32      v4u;
typedef __attribute__((ext_vector_type(8)))  int      v8i;
typedef __attribute__((ext_vector_type(4)))  int      v4i;

// Hard check on the DEVICE pass only: the gfx1250 builtins are not declared for
// the host pass, so __has_builtin is false there by design.
#if defined(__HIP_DEVICE_COMPILE__)
#if !__has_builtin(__builtin_amdgcn_tensor_load_to_lds)
#error "gfx1250 TDM builtin __builtin_amdgcn_tensor_load_to_lds not available (device pass)"
#endif
#if !__has_builtin(__builtin_amdgcn_s_wait_tensorcnt)
#error "gfx1250 builtin __builtin_amdgcn_s_wait_tensorcnt not available (device pass)"
#endif
#endif

__device__ __forceinline__ float dsig(float x) { return 1.f / (1.f + __expf(-x)); }

// ---------------- K0: init workspace + output (every call; ws is poisoned once) --
__global__ void k_init(float* out, u32* hist, u32* meta, u32* idxb) {
    int i = blockIdx.x * blockDim.x + threadIdx.x;
    int st = gridDim.x * blockDim.x;
    for (int j = i; j < O_TOT; j += st)        out[j]  = 0.f;
    for (int j = i; j < 24 * BINS; j += st)    hist[j] = 0u;
    for (int j = i; j < 24 * 4; j += st)       meta[j] = 0u;
    for (int j = i; j < NIMG * POST; j += st)  idxb[j] = 0xFFFFFFFFu;
}

// ---------------- K1: per-level score histogram ---------------------------------
__global__ void k_hist(const float* __restrict__ logits, const float* __restrict__ ctr,
                       u32* __restrict__ hist, int level, int HW) {
    __shared__ u32 sh[BINS];
    int tid = threadIdx.x;
    for (int b = tid; b < BINS; b += blockDim.x) sh[b] = 0u;
    __syncthreads();
    int img = blockIdx.y;
    int total = NCLS * HW;
    int base = blockIdx.x * (256 * 16);
    const float* lg = logits + (size_t)img * total;
    const float* ct = ctr + (size_t)img * HW;
    for (int it = 0; it < 16; ++it) {
        int idx = base + it * 256 + tid;
        if (idx < total) {
            __builtin_prefetch(lg + idx + 4096, 0, 0);   // global_prefetch_b8
            float s = dsig(lg[idx]);
            if (s > PRETH) {
                int pos = idx % HW;
                float f = s * dsig(ct[pos]);
                int bin = (int)(f * (float)BINS);
                if (bin > BINS - 1) bin = BINS - 1;
                atomicAdd(&sh[bin], 1u);
            }
        }
    }
    __syncthreads();
    u32* gh = hist + (img * NLVL + level) * BINS;
    for (int b = tid; b < BINS; b += blockDim.x) {
        u32 v = sh[b];
        if (v) atomicAdd(&gh[b], v);
    }
}

// ---------------- K2: WMMA histogram reduction + threshold bin ------------------
__global__ void k_thresh(const u32* __restrict__ hist, uint4* __restrict__ meta) {
    __shared__ float red[32];
    int il = blockIdx.x;
    int lane = threadIdx.x;          // 32 threads = one wave, EXEC all ones
    const u32* h = hist + il * BINS;

    v8f acc = {};
    v16h ones;
#pragma unroll
    for (int k = 0; k < 16; ++k) ones[k] = (_Float16)1.0f;
#pragma unroll
    for (int it = 0; it < BINS / 512; ++it) {
        v16h a;
        int b0 = it * 512 + lane * 16;
#pragma unroll
        for (int k = 0; k < 16; ++k)
            a[k] = (_Float16)((float)h[b0 + k] * 0.015625f);  // /64 to fit f16 range
        // D = A x ones + C : every D column holds the A row-sums
        acc = __builtin_amdgcn_wmma_f32_16x16x32_f16(false, a, false, ones,
                                                     (short)0, acc, false, false);
    }
    float p = 0.f;
#pragma unroll
    for (int v = 0; v < 8; ++v) p += acc[v];
    red[lane] = p;
    __syncthreads();
    if (lane == 0) {
        float tot = 0.f;
        for (int l = 0; l < 32; ++l) tot += red[l];
        tot = tot * (64.f / 16.f);   // undo scale; D replicates row-sums over 16 cols
        u32 accu = 0; int t = 0;
        for (int b = BINS - 1; b >= 0; --b) {
            accu += h[b];
            if (accu >= TOPK) { t = b; break; }
        }
        meta[il] = make_uint4((u32)t, __float_as_uint(tot), 0u, 0u);
    }
}

// ---------------- K3: compact candidates above threshold bin --------------------
__global__ void k_compact(const float* __restrict__ logits, const float* __restrict__ ctr,
                          u32* __restrict__ meta, uint2* __restrict__ cbuf,
                          int level, int HW) {
    int tid = threadIdx.x;
    int img = blockIdx.y;
    int il = img * NLVL + level;
    float fth = (float)meta[il * 4 + 0] * (1.0f / (float)BINS);
    int total = NCLS * HW;
    int base = blockIdx.x * (256 * 16);
    const float* lg = logits + (size_t)img * total;
    const float* ct = ctr + (size_t)img * HW;
    for (int it = 0; it < 16; ++it) {
        int idx = base + it * 256 + tid;
        if (idx < total) {
            float s = dsig(lg[idx]);
            if (s > PRETH) {
                int pos = idx % HW;
                int cls = idx / HW;
                float f = s * dsig(ct[pos]);
                if (f >= fth) {
                    u32 slot = atomicAdd(&meta[il * 4 + 2], 1u);
                    if (slot < CAP)
                        cbuf[(size_t)il * CAP + slot] =
                            make_uint2(__float_as_uint(f), ((u32)pos << 7) | (u32)cls);
                }
            }
        }
    }
}

// ---------------- bitonic sort (descending), power-of-two n --------------------
__device__ __forceinline__ void bitonic_desc(u64* key, int n, int tid, int nth) {
    for (int k = 2; k <= n; k <<= 1) {
        for (int j = k >> 1; j > 0; j >>= 1) {
            for (int i = tid; i < n; i += nth) {
                int ixj = i ^ j;
                if (ixj > i) {
                    u64 a = key[i], b = key[ixj];
                    bool desc = ((i & k) == 0);
                    if (desc ? (a < b) : (a > b)) { key[i] = b; key[ixj] = a; }
                }
            }
            __syncthreads();
        }
    }
}

// ---------------- K4: per (img,level) sort + box decode -------------------------
__global__ void __launch_bounds__(1024)
k_sortlevel(const u32* __restrict__ meta, const uint2* __restrict__ cbuf,
            const float* loc0, const float* loc1, const float* loc2,
            const float* reg0, const float* reg1, const float* reg2,
            float* __restrict__ rec) {
    __shared__ u64 key[CAP];
    int il = blockIdx.x, img = il / NLVL, level = il % NLVL;
    int tid = threadIdx.x, nth = blockDim.x;
    const int HWs[3] = {15200, 3800, 950};
    const float strd[3] = {8.f, 16.f, 32.f};
    const float* loc = (level == 0) ? loc0 : (level == 1) ? loc1 : loc2;
    const float* rgp = (level == 0) ? reg0 : (level == 1) ? reg1 : reg2;
    int HW = HWs[level];
    u32 cnt = meta[il * 4 + 2]; if (cnt > CAP) cnt = CAP;
    int ncand = (int)__uint_as_float(meta[il * 4 + 1]); if (ncand > TOPK) ncand = TOPK;

    for (int i = tid; i < CAP; i += nth) {
        if (i < (int)cnt) {
            uint2 e = cbuf[(size_t)il * CAP + i];
            key[i] = ((u64)e.x << 32) | e.y;   // scores > 0 => bits monotonic
        } else key[i] = 0ull;
    }
    __syncthreads();
    bitonic_desc(key, CAP, tid, nth);
    __syncthreads();

    if (tid < TOPK) {
        u64 kv = key[tid];
        u32 pk = (u32)kv;
        int cls = (int)(pk & 127u);
        int pos = (int)(pk >> 7);
        float f = __uint_as_float((u32)(kv >> 32));
        bool valid = (tid < (int)cnt) && (tid < ncand) && (kv != 0ull);
        float s = valid ? f : -1.f;
        float xl = loc[2 * pos], yl = loc[2 * pos + 1];
        float sv = strd[level];
        const float* rp = rgp + (size_t)img * 4 * HW + pos;
        float r0 = rp[0] * sv, r1 = rp[HW] * sv, r2 = rp[2 * HW] * sv, r3 = rp[3 * HW] * sv;
        float x1 = fminf(fmaxf(xl - r0, 0.f), IMGW);
        float y1 = fminf(fmaxf(yl - r1, 0.f), IMGH);
        float x2 = fminf(fmaxf(xl + r2, 0.f), IMGW);
        float y2 = fminf(fmaxf(yl + r3, 0.f), IMGH);
        float* o = rec + ((size_t)img * PADK + level * TOPK + tid) * 8;
        o[0] = x1; o[1] = y1; o[2] = x2; o[3] = y2;
        o[4] = s;  o[5] = (float)cls;
        o[6] = __uint_as_float(((u32)level << 16) | (u32)pos);
        o[7] = valid ? 1.f : 0.f;
    }
}

// ---------------- K5: per-image global sort + class-offset NMS boxes ------------
__global__ void __launch_bounds__(1024)
k_sortimage(const float* __restrict__ rec, float* __restrict__ srec,
            float4* __restrict__ nmsb) {
    __shared__ u64 key[SORTN];
    int img = blockIdx.x, tid = threadIdx.x, nth = blockDim.x;
    for (int i = tid; i < SORTN; i += nth) {
        if (i < KTOT) {
            u32 u = __float_as_uint(rec[((size_t)img * PADK + i) * 8 + 4]);
            u32 m = (u >> 31) ? ~u : (u | 0x80000000u);   // total-order float key
            key[i] = ((u64)m << 32) | (u32)i;
        } else key[i] = (u64)(u32)i;                       // pads sort last
    }
    __syncthreads();
    bitonic_desc(key, SORTN, tid, nth);
    __syncthreads();
    for (int r = tid; r < PADK; r += nth) {
        float* o = srec + ((size_t)img * PADK + r) * 8;
        float4 nb;
        if (r < KTOT) {
            int src = (int)(key[r] & 0xFFFFFFFFull);
            const float* s = rec + ((size_t)img * PADK + src) * 8;
            float b0 = s[0], b1 = s[1], b2 = s[2], b3 = s[3];
            float sc = s[4], cf = s[5], gx = s[6], vf = s[7];
            o[0] = b0; o[1] = b1; o[2] = b2; o[3] = b3;
            o[4] = sc; o[5] = cf; o[6] = gx; o[7] = vf;
            if (vf != 0.f) {
                float off = cf * CLSOFF;
                nb = make_float4(b0 + off, b1 + off, b2 + off, b3 + off);
            } else nb = make_float4(-3e8f, -3e8f, -3e8f, -3e8f);
        } else {
            for (int q = 0; q < 8; ++q) o[q] = (q == 4) ? -1.f : 0.f;
            nb = make_float4(-3e8f, -3e8f, -3e8f, -3e8f);
        }
        nmsb[(size_t)img * PADK + r] = nb;
    }
}

// ---------------- K6: TDM load boxes -> LDS, greedy NMS, select top-100 ---------
__global__ void __launch_bounds__(1024)
k_nms(const float* __restrict__ srec, const float4* __restrict__ nmsb,
      float* __restrict__ out, u32* __restrict__ idxb) {
    __shared__ float4 box[PADK];
    __shared__ u32 keep[PADK / 32];
    __shared__ u32 wcnt[PADK / 32];
    __shared__ u32 wpre[PADK / 32 + 1];
    int img = blockIdx.x, tid = threadIdx.x;

#if __has_builtin(__builtin_amdgcn_tensor_load_to_lds)
    // Device pass: single wave issues the TDM DMA (EXEC all ones in this wave).
    if (tid < 32) {
        u64 ga = (u64)(uintptr_t)(nmsb + (size_t)img * PADK);
        u32 lds = (u32)(uintptr_t)(void*)box;          // low 32 bits = LDS offset
        const u32 ndw = PADK * 4;                      // 12288 dwords = 48KB tile
        v4u g0; g0[0] = 1u;                            // count=1, valid D#
        g0[1] = lds;
        g0[2] = (u32)ga;
        g0[3] = ((u32)(ga >> 32) & 0x1FFFFFFu) | (2u << 30);   // type=2 ("image")
        v8i g1 = {0, 0, 0, 0, 0, 0, 0, 0};
        g1[0] = (int)(2u << 16);                       // data_size=4B, no mask
        g1[1] = (int)((ndw & 0xFFFFu) << 16);          // tensor_dim0 lo16 @ [63:48]
        g1[2] = (int)(((ndw >> 16) & 0xFFFFu) | (1u << 16)); // dim0 hi16, dim1=1
        g1[3] = (int)((ndw & 0xFFFFu) << 16);          // tile_dim0 @ [127:112]
        g1[5] = (int)ndw;                              // tensor_dim0_stride lo32
        v4i g2 = {0, 0, 0, 0};
        v4i g3 = {0, 0, 0, 0};
#if defined(__clang_major__) && __clang_major__ >= 23
        v8i g4 = {0, 0, 0, 0, 0, 0, 0, 0};
        __builtin_amdgcn_tensor_load_to_lds(g0, g1, g2, g3, g4, 0);
#else
        __builtin_amdgcn_tensor_load_to_lds(g0, g1, g2, g3, 0);
#endif
        __builtin_amdgcn_s_wait_tensorcnt(0);
    }
#else
    // Host-pass parse fallback (never codegen'd for gfx1250).
    for (int j = tid; j < PADK; j += 1024) box[j] = nmsb[(size_t)img * PADK + j];
#endif
    if (tid < PADK / 32) keep[tid] = 0u;
    __syncthreads();

    // init keep mask from validity flags
    for (int j = tid; j < PADK; j += 1024) {
        float vf = srec[((size_t)img * PADK + j) * 8 + 7];
        if (vf != 0.f) atomicOr(&keep[j >> 5], 1u << (j & 31));
    }
    __syncthreads();

    // exact greedy suppression in score order
    for (int i = 0; i < PADK - 1; ++i) {
        if ((keep[i >> 5] >> (i & 31)) & 1u) {
            float4 bi = box[i];
            float ai = (bi.z - bi.x) * (bi.w - bi.y);
            for (int j = i + 1 + tid; j < PADK; j += 1024) {
                if ((keep[j >> 5] >> (j & 31)) & 1u) {
                    float4 bj = box[j];
                    float xx1 = fmaxf(bi.x, bj.x), yy1 = fmaxf(bi.y, bj.y);
                    float xx2 = fminf(bi.z, bj.z), yy2 = fminf(bi.w, bj.w);
                    float w = fmaxf(xx2 - xx1, 0.f), h = fmaxf(yy2 - yy1, 0.f);
                    float inter = w * h;
                    float aj = (bj.z - bj.x) * (bj.w - bj.y);
                    if (inter / (ai + aj - inter + 1e-9f) > NMSTH)
                        atomicAnd(&keep[j >> 5], ~(1u << (j & 31)));
                }
            }
        }
        __syncthreads();
    }

    // rank kept entries (already score-sorted) and emit first 100
    if (tid < PADK / 32) wcnt[tid] = __popc(keep[tid]);
    __syncthreads();
    if (tid == 0) {
        u32 a = 0;
        for (int w = 0; w < PADK / 32; ++w) { wpre[w] = a; a += wcnt[w]; }
        wpre[PADK / 32] = a;
    }
    __syncthreads();
    for (int j = tid; j < PADK; j += 1024) {
        u32 wb = keep[j >> 5];
        if ((wb >> (j & 31)) & 1u) {
            u32 rank = wpre[j >> 5] + __popc(wb & ((1u << (j & 31)) - 1u));
            if (rank < POST) {
                const float* s = srec + ((size_t)img * PADK + j) * 8;
                float* ob = out + O_BOX + ((size_t)img * POST + rank) * 4;
                ob[0] = s[0]; ob[1] = s[1]; ob[2] = s[2]; ob[3] = s[3];
                out[O_SCR + img * POST + rank] = sqrtf(fmaxf(s[4], 1e-12f));
                out[O_CLS + img * POST + rank] = s[5];
                out[O_KV  + img * POST + rank] = 1.f;
                idxb[img * POST + rank] = __float_as_uint(s[6]);
            }
        }
    }
}

// ---------------- K7: gather 128-d cofs for surviving detections ----------------
__global__ void k_cofs(const u32* __restrict__ idxb, const float* c0, const float* c1,
                       const float* c2, float* __restrict__ out) {
    int r = blockIdx.x, img = blockIdx.y, c = threadIdx.x;
    u32 g = idxb[img * POST + r];
    if (g == 0xFFFFFFFFu) return;
    int level = (int)(g >> 16), pos = (int)(g & 0xFFFFu);
    const int HWs[3] = {15200, 3800, 950};
    const float* cp = (level == 0) ? c0 : (level == 1) ? c1 : c2;
    int HW = HWs[level];
    out[O_COF + ((size_t)(img * POST + r)) * COFD + c] =
        cp[((size_t)img * COFD + c) * HW + pos];
}

// -------------------------------------------------------------------------------
extern "C" void kernel_launch(void* const* d_in, const int* in_sizes, int n_in,
                              void* d_out, int out_size, void* d_ws, size_t ws_size,
                              hipStream_t stream) {
    (void)in_sizes; (void)n_in; (void)out_size; (void)ws_size;
    char* ws = (char*)d_ws;
    u32*    hist = (u32*)(ws + WS_HIST);
    u32*    meta = (u32*)(ws + WS_META);
    uint2*  cbuf = (uint2*)(ws + WS_CBUF);
    float*  rec  = (float*)(ws + WS_REC);
    float*  srec = (float*)(ws + WS_SREC);
    float4* nmsb = (float4*)(ws + WS_NMSB);
    u32*    idxb = (u32*)(ws + WS_IDX);
    float*  out  = (float*)d_out;

    const int HWs[3] = {15200, 3800, 950};

    k_init<<<256, 256, 0, stream>>>(out, hist, meta, idxb);

    for (int l = 0; l < 3; ++l) {
        int total = NCLS * HWs[l];
        int gx = (total + 4095) / 4096;
        k_hist<<<dim3(gx, NIMG), 256, 0, stream>>>(
            (const float*)d_in[l * 5 + 1], (const float*)d_in[l * 5 + 3],
            hist, l, HWs[l]);
    }

    k_thresh<<<24, 32, 0, stream>>>(hist, (uint4*)meta);

    for (int l = 0; l < 3; ++l) {
        int total = NCLS * HWs[l];
        int gx = (total + 4095) / 4096;
        k_compact<<<dim3(gx, NIMG), 256, 0, stream>>>(
            (const float*)d_in[l * 5 + 1], (const float*)d_in[l * 5 + 3],
            meta, cbuf, l, HWs[l]);
    }

    k_sortlevel<<<24, 1024, 0, stream>>>(
        meta, cbuf,
        (const float*)d_in[0], (const float*)d_in[5], (const float*)d_in[10],
        (const float*)d_in[2], (const float*)d_in[7], (const float*)d_in[12],
        rec);

    k_sortimage<<<NIMG, 1024, 0, stream>>>(rec, srec, nmsb);

    k_nms<<<NIMG, 1024, 0, stream>>>(srec, nmsb, out, idxb);

    k_cofs<<<dim3(POST, NIMG), COFD, 0, stream>>>(
        idxb, (const float*)d_in[4], (const float*)d_in[9], (const float*)d_in[14],
        out);
}
